// ABCNN_58695023067400
// MI455X (gfx1250) — matrix operand order repaired and checked by reading, not compiled, not run on hardware
//
#include <hip/hip_runtime.h>

typedef __attribute__((ext_vector_type(16))) _Float16 v16h;
typedef __attribute__((ext_vector_type(8)))  float    v8f;

#define NB   2048
#define SL   32
#define SD   256
#define SLP  34   // L+2

// ---- LDS layout (float slots). rep1 region reused for c2; att1 region reused for att2.
#define OFF_R0    0        // 8704 : rep1 (8192 used), later c2 (34x256)
#define OFF_REP2  8704     // 8192 : rep2
#define OFF_ATT   16896    // 8192 : att1, later att2
#define OFF_C1    25088    // 8704 : c1 (34x256)
#define OFF_A1    33792    // 1024 : A1 (32x32)
#define OFF_A2    34816    // 1156 : A2 (34x34)
#define OFF_A2N   35972    // 32
#define OFF_B2N   36004    // 32
#define OFF_C1N   36036    // 34
#define OFF_C2N   36070    // 34
#define OFF_COL   36104    // 34
#define OFF_ROW   36138    // 34
#define OFF_V12   36172    // 512
#define OFF_H     36684    // 64
#define OFF_SIDX  36748    // 64 ints
#define SMEM_FLOATS 36812  // ~144 KB -> 2 workgroups / WGP (320 KB LDS)

// A fragment (16x32 f16) from an f32 row-major matrix, rows >= mlim read as 0.
__device__ __forceinline__ v16h frag_a_rows(const float* M, int ld, int m0, int k0,
                                            int lane, int mlim) {
  int m = m0 + (lane & 15);
  int h = lane >> 4;
  bool ok = (m < mlim);
  const float* row = M + m * ld;
  v16h a;
#pragma unroll
  for (int e = 0; e < 16; ++e) {
    int k = k0 + e + 16 * h;
    a[e] = (_Float16)(ok ? row[k] : 0.0f);
  }
  return a;
}

// B fragment (32x16 f16) where B[k][n] = M[n0+n][k0+k]  (B = M^T, M row-major f32)
__device__ __forceinline__ v16h frag_bT_rows(const float* M, int ld, int n0, int k0,
                                             int lane, int nlim) {
  int n = n0 + (lane & 15);
  int h = lane >> 4;
  bool ok = (n < nlim);
  const float* row = M + n * ld;
  v16h b;
#pragma unroll
  for (int e = 0; e < 16; ++e) {
    int k = k0 + e + 16 * h;
    b[e] = (_Float16)(ok ? row[k] : 0.0f);
  }
  return b;
}

// B fragment (32x16 f16) where B[k][n] = W[(k0+k)*ld + n0+n]  (W row-major f32, global)
__device__ __forceinline__ v16h frag_b_w(const float* W, int ld, int k0, int n0, int lane) {
  int n = n0 + (lane & 15);
  int h = lane >> 4;
  v16h b;
#pragma unroll
  for (int e = 0; e < 16; ++e) {
    int k = k0 + e + 16 * h;
    b[e] = (_Float16)W[k * ld + n];
  }
  return b;
}

#define WMMA_F32_F16(A, Bf, C) \
  __builtin_amdgcn_wmma_f32_16x16x32_f16(false, (A), false, (Bf), (short)0, (C), false, false)

extern "C" __global__ __launch_bounds__(256)
void abcnn_main(const int* __restrict__ s1, const int* __restrict__ s2,
                const float* __restrict__ embed,
                const float* __restrict__ c1w, const float* __restrict__ c1b,
                const float* __restrict__ c2w, const float* __restrict__ c2b,
                const float* __restrict__ c3w, const float* __restrict__ c3b,
                const float* __restrict__ W0, const float* __restrict__ W1,
                const float* __restrict__ fw1, const float* __restrict__ fb1,
                const float* __restrict__ fw2, const float* __restrict__ fb2,
                float* __restrict__ logits /* NB x 2, pre-softmax */) {
  extern __shared__ float smem[];
  float* rep1 = smem + OFF_R0;
  float* c2m  = smem + OFF_R0;     // aliases rep1 (dead by then)
  float* rep2 = smem + OFF_REP2;
  float* att  = smem + OFF_ATT;    // att1 then att2
  float* c1m  = smem + OFF_C1;
  float* A1m  = smem + OFF_A1;
  float* A2m  = smem + OFF_A2;
  float* a2n  = smem + OFF_A2N;
  float* b2n  = smem + OFF_B2N;
  float* c1n  = smem + OFF_C1N;
  float* c2n  = smem + OFF_C2N;
  float* colS = smem + OFF_COL;
  float* rowS = smem + OFF_ROW;
  float* v12  = smem + OFF_V12;
  float* hbuf = smem + OFF_H;
  int*   sidx = (int*)(smem + OFF_SIDX);

  const int b    = blockIdx.x;
  const int tid  = threadIdx.x;
  const int lane = tid & 31;
  const int wv   = tid >> 5;

  // ---- scalar weights (uniform -> SGPR loads)
  const float w10 = c1w[0], w11 = c1w[1], w12 = c1w[2], bias1 = c1b[0];
  const float cw0 = w10, cw1 = w10 + w11, cw2 = w10 + w11 + w12, cw3 = w11 + w12, cw4 = w12;
  const float w3a0 = c3w[0], w3a1 = c3w[1], w3a2 = c3w[2];
  const float w3b0 = c3w[3], w3b1 = c3w[4], w3b2 = c3w[5];
  const float bias3 = c3b[0];
  const float Wc2 = c2w[0] + c2w[1] + c2w[2], bias2 = c2b[0];

  if (tid < SL) {
    sidx[tid]      = s1[b * SL + tid];
    sidx[SL + tid] = s2[b * SL + tid];
  }
  __syncthreads();

  // ---- Phase 1: fused embed-gather + conv_h + avgpool3 -> rep1, rep2.
  // rep[i][d] = (1/3) * sum_{delta=-2..2} cw[delta+2] * e[i+delta][d] + bias1
  {
    const int d = tid;
    float g1[SL], g2[SL];
#pragma unroll
    for (int j = 0; j < SL; ++j) {
      g1[j] = embed[(size_t)sidx[j] * SD + d];
      g2[j] = embed[(size_t)sidx[SL + j] * SD + d];
    }
    const float cw[5] = {cw0, cw1, cw2, cw3, cw4};
#pragma unroll
    for (int i = 0; i < SL; ++i) {
      float acc1 = 0.0f, acc2 = 0.0f;
#pragma unroll
      for (int dl = -2; dl <= 2; ++dl) {
        int j = i + dl;
        if (j >= 0 && j < SL) {
          acc1 += cw[dl + 2] * g1[j];
          acc2 += cw[dl + 2] * g2[j];
        }
      }
      rep1[i * SD + d] = acc1 * (1.0f / 3.0f) + bias1;
      rep2[i * SD + d] = acc2 * (1.0f / 3.0f) + bias1;
    }
  }
  __syncthreads();

  // ---- Phase 2: squared row norms (f32, deterministic)
  if (tid < SL) {
    float s = 0.0f;
    for (int d = 0; d < SD; ++d) { float x = rep1[tid * SD + d]; s += x * x; }
    a2n[tid] = s;
  } else if (tid < 2 * SL) {
    int i = tid - SL;
    float s = 0.0f;
    for (int d = 0; d < SD; ++d) { float x = rep2[i * SD + d]; s += x * x; }
    b2n[i] = s;
  }
  __syncthreads();

  // ---- Phase 3: Gram G1 = rep1 @ rep2^T via WMMA, then A1 = 1/(1+dist)
  if (wv < 4) {
    const int mt = (wv >> 1) * 16, nt = (wv & 1) * 16;
    v8f acc = {};
    for (int k0 = 0; k0 < SD; k0 += 32) {
      v16h af = frag_a_rows(rep1, SD, mt, k0, lane, SL);
      v16h bf = frag_bT_rows(rep2, SD, nt, k0, lane, SL);
      acc = WMMA_F32_F16(af, bf, acc);
    }
    const int h = lane >> 4, n = lane & 15;
#pragma unroll
    for (int r = 0; r < 8; ++r) {
      int i = mt + r + 8 * h, j = nt + n;
      float d2 = a2n[i] + b2n[j] - 2.0f * acc[r];
      A1m[i * SL + j] = 1.0f / (1.0f + sqrtf(fmaxf(d2, 0.0f)));
    }
  }
  __syncthreads();

  // ---- Phase 4: att1 = A1 @ W0  (32 tiles, 1 WMMA each, K=32)
  for (int t = wv; t < 32; t += 8) {
    const int mt = (t >> 4) * 16, nt = (t & 15) * 16;
    v16h af = frag_a_rows(A1m, SL, mt, 0, lane, SL);
    v16h bf = frag_b_w(W0, SD, 0, nt, lane);
    v8f acc = {};
    acc = WMMA_F32_F16(af, bf, acc);
    const int h = lane >> 4, n = lane & 15;
#pragma unroll
    for (int r = 0; r < 8; ++r) att[(mt + r + 8 * h) * SD + nt + n] = acc[r];
  }
  __syncthreads();

  // ---- Phase 5: c1 = conv3d_21(stack(rep1, att1))
  {
    const int d = tid;
    for (int t = 0; t < SLP; ++t) {
      float acc = bias3;
      int s;
      s = t - 2; if (s >= 0 && s < SL) acc += w3a0 * rep1[s * SD + d] + w3b0 * att[s * SD + d];
      s = t - 1; if (s >= 0 && s < SL) acc += w3a1 * rep1[s * SD + d] + w3b1 * att[s * SD + d];
      s = t;     if (s < SL)           acc += w3a2 * rep1[s * SD + d] + w3b2 * att[s * SD + d];
      c1m[t * SD + d] = acc;
    }
  }
  __syncthreads();

  // ---- Phase 6: att2 = A1 @ W1 (overwrites att1; c1 already materialized)
  for (int t = wv; t < 32; t += 8) {
    const int mt = (t >> 4) * 16, nt = (t & 15) * 16;
    v16h af = frag_a_rows(A1m, SL, mt, 0, lane, SL);
    v16h bf = frag_b_w(W1, SD, 0, nt, lane);
    v8f acc = {};
    acc = WMMA_F32_F16(af, bf, acc);
    const int h = lane >> 4, n = lane & 15;
#pragma unroll
    for (int r = 0; r < 8; ++r) att[(mt + r + 8 * h) * SD + nt + n] = acc[r];
  }
  __syncthreads();

  // ---- Phase 7: c2 = conv3d_21(stack(rep2, att2))  (written over dead rep1)
  {
    const int d = tid;
    for (int t = 0; t < SLP; ++t) {
      float acc = bias3;
      int s;
      s = t - 2; if (s >= 0 && s < SL) acc += w3a0 * rep2[s * SD + d] + w3b0 * att[s * SD + d];
      s = t - 1; if (s >= 0 && s < SL) acc += w3a1 * rep2[s * SD + d] + w3b1 * att[s * SD + d];
      s = t;     if (s < SL)           acc += w3a2 * rep2[s * SD + d] + w3b2 * att[s * SD + d];
      c2m[t * SD + d] = acc;
    }
  }
  __syncthreads();

  // ---- Phase 8: row norms of c1, c2
  if (tid < SLP) {
    float s = 0.0f;
    for (int d = 0; d < SD; ++d) { float x = c1m[tid * SD + d]; s += x * x; }
    c1n[tid] = s;
  } else if (tid >= 64 && tid < 64 + SLP) {
    int i = tid - 64;
    float s = 0.0f;
    for (int d = 0; d < SD; ++d) { float x = c2m[i * SD + d]; s += x * x; }
    c2n[i] = s;
  }
  __syncthreads();

  // ---- Phase 9: Gram G2 = c1 @ c2^T (34x34 via 3x3 tiles), A2 = 1/(1+dist)
  for (int t = wv; t < 9; t += 8) {
    const int mt = (t / 3) * 16, nt = (t % 3) * 16;
    v8f acc = {};
    for (int k0 = 0; k0 < SD; k0 += 32) {
      v16h af = frag_a_rows(c1m, SD, mt, k0, lane, SLP);
      v16h bf = frag_bT_rows(c2m, SD, nt, k0, lane, SLP);
      acc = WMMA_F32_F16(af, bf, acc);
    }
    const int h = lane >> 4, n = lane & 15;
#pragma unroll
    for (int r = 0; r < 8; ++r) {
      int i = mt + r + 8 * h, j = nt + n;
      if (i < SLP && j < SLP) {
        float d2 = c1n[i] + c2n[j] - 2.0f * acc[r];
        A2m[i * SLP + j] = 1.0f / (1.0f + sqrtf(fmaxf(d2, 0.0f)));
      }
    }
  }
  __syncthreads();

  // ---- Phase 10: col/row sums of A2
  if (tid < SLP) {
    float s = 0.0f;
    for (int j = 0; j < SLP; ++j) s += A2m[tid * SLP + j];
    colS[tid] = s;
  } else if (tid >= 64 && tid < 64 + SLP) {
    int j = tid - 64;
    float s = 0.0f;
    for (int i = 0; i < SLP; ++i) s += A2m[i * SLP + j];
    rowS[j] = s;
  }
  __syncthreads();

  // ---- Phase 11: v1/v2 = mean(conv_h(avgpool3(c * scale))) collapsed to weighted row-sum
  {
    const int d = tid;
    float s1v = 0.0f, s2v = 0.0f;
    for (int t = 0; t < SLP; ++t) {
      float m = (t <= 1) ? (float)(t + 1) : ((t >= SL) ? (float)(SLP - t) : 3.0f);
      s1v += m * c1m[t * SD + d] * colS[t];
      s2v += m * c2m[t * SD + d] * rowS[t];
    }
    const float sc = Wc2 / (3.0f * (float)SLP);
    v12[d]      = s1v * sc + bias2;
    v12[SD + d] = s2v * sc + bias2;
  }
  __syncthreads();

  // ---- Phase 12: MLP head
  if (tid < 64) {
    float acc = fb1[tid];
    for (int q = 0; q < 2 * SD; ++q) acc += v12[q] * fw1[q * 64 + tid];
    hbuf[tid] = tanhf(acc);
  }
  __syncthreads();
  if (tid < 2) {
    float acc = fb2[tid];
    for (int o = 0; o < 64; ++o) acc += hbuf[o] * fw2[o * 2 + tid];
    logits[b * 2 + tid] = acc;
  }
}

// Softmax over the BATCH axis (dim 0), per column, in place on d_out.
extern "C" __global__ __launch_bounds__(256)
void abcnn_softmax(float* __restrict__ out) {
  __shared__ float red0[256], red1[256];
  const int t = threadIdx.x;
  float l0[8], l1[8];
  float mx0 = -1e30f, mx1 = -1e30f;
#pragma unroll
  for (int r = 0; r < 8; ++r) {
    int bb = t + r * 256;
    l0[r] = out[bb * 2 + 0];
    l1[r] = out[bb * 2 + 1];
    mx0 = fmaxf(mx0, l0[r]);
    mx1 = fmaxf(mx1, l1[r]);
  }
  red0[t] = mx0; red1[t] = mx1;
  __syncthreads();
  for (int s = 128; s > 0; s >>= 1) {
    if (t < s) { red0[t] = fmaxf(red0[t], red0[t + s]); red1[t] = fmaxf(red1[t], red1[t + s]); }
    __syncthreads();
  }
  mx0 = red0[0]; mx1 = red1[0];
  __syncthreads();
  float s0 = 0.0f, s1 = 0.0f;
#pragma unroll
  for (int r = 0; r < 8; ++r) {
    l0[r] = expf(l0[r] - mx0); s0 += l0[r];
    l1[r] = expf(l1[r] - mx1); s1 += l1[r];
  }
  red0[t] = s0; red1[t] = s1;
  __syncthreads();
  for (int s = 128; s > 0; s >>= 1) {
    if (t < s) { red0[t] += red0[t + s]; red1[t] += red1[t + s]; }
    __syncthreads();
  }
  s0 = red0[0]; s1 = red1[0];
#pragma unroll
  for (int r = 0; r < 8; ++r) {
    int bb = t + r * 256;
    out[bb * 2 + 0] = l0[r] / s0;
    out[bb * 2 + 1] = l1[r] / s1;
  }
}

extern "C" void kernel_launch(void* const* d_in, const int* in_sizes, int n_in,
                              void* d_out, int out_size, void* d_ws, size_t ws_size,
                              hipStream_t stream) {
  (void)in_sizes; (void)n_in; (void)out_size; (void)d_ws; (void)ws_size;
  const int*   s1    = (const int*)d_in[0];
  const int*   s2    = (const int*)d_in[1];
  const float* embed = (const float*)d_in[2];
  const float* c1w   = (const float*)d_in[3];
  const float* c1b   = (const float*)d_in[4];
  const float* c2w   = (const float*)d_in[5];
  const float* c2b   = (const float*)d_in[6];
  const float* c3w   = (const float*)d_in[7];
  const float* c3b   = (const float*)d_in[8];
  const float* W0    = (const float*)d_in[9];
  const float* W1    = (const float*)d_in[10];
  const float* fw1   = (const float*)d_in[11];
  const float* fb1   = (const float*)d_in[12];
  const float* fw2   = (const float*)d_in[13];
  const float* fb2   = (const float*)d_in[14];
  float* out = (float*)d_out;

  const size_t smem = (size_t)SMEM_FLOATS * sizeof(float);
  abcnn_main<<<NB, 256, smem, stream>>>(s1, s2, embed, c1w, c1b, c2w, c2b, c3w, c3b,
                                        W0, W1, fw1, fb1, fw2, fb2, out);
  abcnn_softmax<<<1, 256, 0, stream>>>(out);
}
